// MultiPrecisionLinear_68496138437350
// MI455X (gfx1250) — matrix operand order repaired and checked by reading, not compiled, not run on hardware
//
#include <hip/hip_runtime.h>

typedef __attribute__((ext_vector_type(16))) __bf16        v16bf;
typedef __attribute__((ext_vector_type(8)))  float         v8f;
typedef __attribute__((ext_vector_type(4)))  unsigned int  v4u;
typedef __attribute__((ext_vector_type(8)))  unsigned int  v8u;

#define DIN        256
#define DOUT       256
#define MP         32768     // rows per path = bpp(8) * S(4096)
#define MTILE      128       // rows per block (8 waves x 16)
#define KCHUNK     32        // K per bf16 WMMA step
#define LDS_STRIDE 20        // u32 per LDS row: 16 data (32 bf16) + 4 pad (80B rows)

// pack {bf16_trunc(f1), bf16_trunc(f0)} into one dword with a single v_perm_b32
__device__ __forceinline__ unsigned pack_bf16_hi2(float f0, float f1) {
  return __builtin_amdgcn_perm(__float_as_uint(f1), __float_as_uint(f0), 0x07060302u);
}
__device__ __forceinline__ float bf16_back(float f) {
  return __uint_as_float(__float_as_uint(f) & 0xFFFF0000u);
}

// convert 2 f32 -> packed hi dword + packed lo(residual) dword
__device__ __forceinline__ void split2(float f0, float f1, unsigned& hi, unsigned& lo) {
  hi = pack_bf16_hi2(f0, f1);
  float r0 = f0 - bf16_back(f0);
  float r1 = f1 - bf16_back(f1);
  lo = pack_bf16_hi2(r0, r1);
}

__global__ __launch_bounds__(256)
void mp_linear_wmma(const float* __restrict__ x,
                    const float* __restrict__ weight_bank,
                    const float* __restrict__ bias,
                    const int*   __restrict__ assigned_bits,
                    float* __restrict__ out) {
  __shared__ unsigned ldsWhi[DOUT * LDS_STRIDE];   // 20 KB
  __shared__ unsigned ldsWlo[DOUT * LDS_STRIDE];   // 20 KB

  const int p    = blockIdx.y;
  const int tid  = threadIdx.x;
  const int lane = tid & 31;
  const int wave = tid >> 5;
  const int half = lane >> 4;    // 0: lanes 0-15, 1: lanes 16-31
  const int l16  = lane & 15;

  const float* xp = x   + (size_t)p * MP * DIN;
  float*       op = out + (size_t)p * MP * DOUT;
  const float* Wp = weight_bank + (size_t)assigned_bits[p] * (DOUT * DIN);

  const int m0 = blockIdx.x * MTILE + wave * 16;
  const float* arow = xp + (size_t)(m0 + l16) * DIN;   // lanes l and l+16 share a row

  v8f acc[16];
  #pragma unroll
  for (int i = 0; i < 16; ++i) acc[i] = v8f{};

  // B-tile LDS read base for this lane: row = (nt*16 + l16), column-half by lane half
  const unsigned* bbase_hi = ldsWhi + (size_t)l16 * LDS_STRIDE + half * 4;
  const unsigned* bbase_lo = ldsWlo + (size_t)l16 * LDS_STRIDE + half * 4;

  #pragma unroll 1
  for (int kc = 0; kc < DIN / KCHUNK; ++kc) {
    const int k0 = kc * KCHUNK;

    // ---- cooperative stage: W[:, k0:k0+32] -> packed bf16 hi/lo planes in LDS ----
    {
      const float4* wrow = (const float4*)(Wp + (size_t)tid * DIN + k0);
      unsigned* dhi = ldsWhi + tid * LDS_STRIDE;
      unsigned* dlo = ldsWlo + tid * LDS_STRIDE;
      #pragma unroll
      for (int j = 0; j < 8; ++j) {
        float4 f = wrow[j];
        unsigned h0, l0, h1, l1;
        split2(f.x, f.y, h0, l0);
        split2(f.z, f.w, h1, l1);
        dhi[j * 2 + 0] = h0;
        dhi[j * 2 + 1] = h1;
        dlo[j * 2 + 0] = l0;
        dlo[j * 2 + 1] = l1;
      }
    }
    __syncthreads();

    // ---- build A operand (16x32 bf16 layout) straight from global f32 ----
    // lanes 0-15 : elems 0..7 = K k0+0..7,  elems 8..15 = K k0+16..23
    // lanes 16-31: elems 0..7 = K k0+8..15, elems 8..15 = K k0+24..31
    v16bf Ahi, Alo;
    {
      const float4* ap0 = (const float4*)(arow + k0 + half * 8);
      const float4* ap1 = (const float4*)(arow + k0 + 16 + half * 8);
      float4 a0 = ap0[0], a1 = ap0[1];
      float4 a2 = ap1[0], a3 = ap1[1];
      float v[16] = {a0.x, a0.y, a0.z, a0.w, a1.x, a1.y, a1.z, a1.w,
                     a2.x, a2.y, a2.z, a2.w, a3.x, a3.y, a3.z, a3.w};
      v8u ah, al;
      #pragma unroll
      for (int e = 0; e < 8; ++e) {
        unsigned h, l;
        split2(v[2 * e], v[2 * e + 1], h, l);
        ah[e] = h;
        al[e] = l;
      }
      Ahi = __builtin_bit_cast(v16bf, ah);
      Alo = __builtin_bit_cast(v16bf, al);
    }

    // ---- 16 N-tiles, software-pipelined B loads from LDS ----
    // B 32x16 layout: lane = column (l16), elems 0..7 = K 8*half+0..7, elems 8..15 = K 16+8*half+0..7
    v16bf Bhi, Blo;
    {
      v4u h0 = *(const v4u*)(bbase_hi);
      v4u h1 = *(const v4u*)(bbase_hi + 8);
      v4u l0 = *(const v4u*)(bbase_lo);
      v4u l1 = *(const v4u*)(bbase_lo + 8);
      Bhi = __builtin_bit_cast(v16bf, __builtin_shufflevector(h0, h1, 0,1,2,3,4,5,6,7));
      Blo = __builtin_bit_cast(v16bf, __builtin_shufflevector(l0, l1, 0,1,2,3,4,5,6,7));
    }
    #pragma unroll
    for (int nt = 0; nt < 16; ++nt) {
      v16bf Bhi_n, Blo_n;
      if (nt < 15) {
        const unsigned* nh = bbase_hi + (nt + 1) * 16 * LDS_STRIDE;
        const unsigned* nl = bbase_lo + (nt + 1) * 16 * LDS_STRIDE;
        v4u h0 = *(const v4u*)(nh);
        v4u h1 = *(const v4u*)(nh + 8);
        v4u l0 = *(const v4u*)(nl);
        v4u l1 = *(const v4u*)(nl + 8);
        Bhi_n = __builtin_bit_cast(v16bf, __builtin_shufflevector(h0, h1, 0,1,2,3,4,5,6,7));
        Blo_n = __builtin_bit_cast(v16bf, __builtin_shufflevector(l0, l1, 0,1,2,3,4,5,6,7));
      }
      acc[nt] = __builtin_amdgcn_wmma_f32_16x16x32_bf16(false, Ahi, false, Bhi,
                                                        (short)0, acc[nt], false, false);
      acc[nt] = __builtin_amdgcn_wmma_f32_16x16x32_bf16(false, Ahi, false, Blo,
                                                        (short)0, acc[nt], false, false);
      acc[nt] = __builtin_amdgcn_wmma_f32_16x16x32_bf16(false, Alo, false, Bhi,
                                                        (short)0, acc[nt], false, false);
      if (nt < 15) { Bhi = Bhi_n; Blo = Blo_n; }
    }
    __syncthreads();
  }

  // ---- epilogue: C/D layout (VGPR v -> M = v + half*8, N = l16), add bias ----
  const int mb = m0 + half * 8;
  #pragma unroll
  for (int nt = 0; nt < 16; ++nt) {
    const int col = nt * 16 + l16;
    const float bv = bias[col];
    #pragma unroll
    for (int v = 0; v < 8; ++v) {
      __builtin_nontemporal_store(acc[nt][v] + bv,
                                  op + (size_t)(mb + v) * DOUT + col);
    }
  }
}

extern "C" void kernel_launch(void* const* d_in, const int* in_sizes, int n_in,
                              void* d_out, int out_size, void* d_ws, size_t ws_size,
                              hipStream_t stream) {
  (void)in_sizes; (void)n_in; (void)out_size; (void)d_ws; (void)ws_size;
  const float* x      = (const float*)d_in[0];
  const float* wbank  = (const float*)d_in[1];
  const float* bias   = (const float*)d_in[2];
  const int*   bits   = (const int*)d_in[3];
  float*       out    = (float*)d_out;

  dim3 grid(MP / MTILE, 8);   // 256 M-tiles per path x 8 paths
  dim3 block(256);            // 8 waves (wave32)
  mp_linear_wmma<<<grid, block, 0, stream>>>(x, wbank, bias, bits, out);
}
